// STConvBlock_23433341567344
// MI455X (gfx1250) — matrix operand order repaired and checked by reading, not compiled, not run on hardware
//
#include <hip/hip_runtime.h>
#include <hip/hip_bf16.h>

// ---------------- problem constants ----------------
#define BB     8
#define NN     2048
#define TT     64
#define C_IN   32
#define C_OUT  64
#define KCHEB  3
#define KT     3
#define EPSV   1e-5f

#define NODES  (BB * NN)            // 16384
#define MTOT   (BB * NN * TT)       // 1048576 rows of width 64
#define JCOLS  (TT * C_OUT)         // 4096 GEMM cols per batch

// ---------------- CDNA5 feature probes (compile-safe) ----------------
#if defined(__has_builtin)
#  if __has_builtin(__builtin_amdgcn_tensor_load_to_lds)
#    define USE_TDM 1
#  elif __has_builtin(__builtin_amdgcn_global_load_async_to_lds_b128)
#    define USE_ASYNC 1
#  endif
#endif

// ---------------- WMMA / TDM types ----------------
typedef __attribute__((ext_vector_type(16))) __bf16 v16bf;
typedef __attribute__((ext_vector_type(8)))  float  v8f;
typedef unsigned __attribute__((ext_vector_type(4))) v4u;
typedef int      __attribute__((ext_vector_type(4))) v4i;
typedef int      __attribute__((ext_vector_type(8))) v8i;

union FragBF {
    v16bf    bf;
    unsigned u[8];
};

__device__ __forceinline__ unsigned short f2bf(float f) {
    unsigned u = __float_as_uint(f);
    unsigned r = (u + 0x7FFFu + ((u >> 16) & 1u)) >> 16;   // RNE
    return (unsigned short)r;
}
__device__ __forceinline__ float bf2f(unsigned short h) {
    return __uint_as_float(((unsigned)h) << 16);
}

#if defined(USE_TDM)
// TDM descriptor: 2D tile 32(k) x 128(m) bf16 out of a 2048-stride tensor,
// HW row padding 4 DWORDs per 16 DWORDs -> 80B LDS pitch (SA=40 u16).
__device__ __forceinline__ void tdm_load_A(const unsigned short* gsrc, unsigned lds_addr) {
    unsigned long long ga = (unsigned long long)(const void*)gsrc;
    v4u g0;
    g0[0] = 1u;                                     // count=1
    g0[1] = lds_addr;                               // lds_addr
    g0[2] = (unsigned)ga;                           // global_addr lo
    g0[3] = (unsigned)((ga >> 32) & 0x01FFFFFFull)  // global_addr hi
            | (2u << 30);                           // type=2 (image)
    v8i g1;
    g1[0] = (1 << 16)      // data_size = 2 bytes
          | (1 << 20)      // pad_enable
          | (3 << 22)      // pad_interval: 16 DWORDs (one 64B row)
          | (3 << 25);     // pad_amount:   4 DWORDs (16B) -> 80B pitch
    g1[1] = (int)(2048u << 16);   // tensor_dim0[15:0]
    g1[2] = (int)(2048u << 16);   // dim0 hi=0 | tensor_dim1[15:0]
    g1[3] = (int)(32u << 16);     // dim1 hi=0 | tile_dim0=32
    g1[4] = 128;                  // tile_dim1=128, tile_dim2=0
    g1[5] = 2048;                 // tensor_dim0_stride lo32
    g1[6] = 0;
    g1[7] = 0;
    v4i gz; gz[0] = 0; gz[1] = 0; gz[2] = 0; gz[3] = 0;
#  if __clang_major__ >= 23
    v8i gz8; for (int q = 0; q < 8; ++q) gz8[q] = 0;
    __builtin_amdgcn_tensor_load_to_lds(g0, g1, gz, gz, gz8, 0);
#  else
    __builtin_amdgcn_tensor_load_to_lds(g0, g1, gz, gz, 0);
#  endif
}
#endif

// =====================================================================
// prep: convert L_tilde -> bf16; build bf16 im2col weight matrices.
// =====================================================================
__global__ void prep_kernel(const float* __restrict__ L,
                            const float* __restrict__ w1,
                            const float* __restrict__ w2,
                            const float* __restrict__ cw,
                            unsigned short* __restrict__ Lbf,
                            unsigned short* __restrict__ W1m,
                            unsigned short* __restrict__ W2m,
                            unsigned short* __restrict__ Wcat) {
    long i = (long)blockIdx.x * blockDim.x + threadIdx.x;
    const long NL = (long)NN * NN;
    if (i < NL) Lbf[i] = f2bf(L[i]);
    if (i < 64 * 96) {
        int t = (int)(i / 96), k = (int)(i % 96);
        int kt = k / C_IN, ci = k % C_IN;
        W1m[i] = f2bf(w1[(t * C_IN + ci) * KT + kt]);
    }
    if (i < 64 * 192) {
        int t = (int)(i / 192), k = (int)(i % 192);
        int kt = k / C_OUT, ci = k % C_OUT;
        W2m[i] = f2bf(w2[(t * C_OUT + ci) * KT + kt]);
    }
    if (i < KCHEB * 64 * 64) Wcat[i] = f2bf(cw[i]);
}

// =====================================================================
// Temporal conv via WMMA (handles the reference's reshape scramble):
//   O[t,c] = sum_k Wm[t,k] * Xim[k,c] + bias[t]
//   Xim[(kt,ci), c] = X[node, c+kt-1, ci] (zero pad in time)
// =====================================================================
template <int CINP, bool OUTBF>
__global__ void tconv_kernel(const float* __restrict__ X,
                             const unsigned short* __restrict__ Wm,
                             const float* __restrict__ bias,
                             unsigned short* __restrict__ OutBf,
                             float* __restrict__ OutF) {
    constexpr int KK = 3 * CINP;      // 96 or 192
    constexpr int KP = KK + 8;        // padded LDS row (even)
    __shared__ unsigned short ldsW[64 * KP];        // A: [t][k]
    __shared__ unsigned short ldsX[2][64 * KP];     // B^T: [c][k] per node

    const int tid = threadIdx.x;
    const int node0 = blockIdx.x * 2;

    for (int i = tid; i < 64 * KK; i += 256) {
        int t = i / KK, k = i % KK;
        ldsW[t * KP + k] = Wm[i];
    }
    for (int g = 0; g < 2; ++g) {
        const float* xn = X + (size_t)(node0 + g) * TT * CINP;
        for (int i = tid; i < 64 * KK; i += 256) {
            int c = i / KK, k = i % KK;
            int kt = k / CINP, ci = k % CINP;
            int tt = c + kt - 1;
            float v = (tt >= 0 && tt < TT) ? xn[tt * CINP + ci] : 0.0f;
            ldsX[g][c * KP + k] = f2bf(v);
        }
    }
    __syncthreads();

    const int wave = tid >> 5, lane = tid & 31;
    const int h = lane >> 4, l15 = lane & 15;
    const int g = wave >> 2;
    const int strip = (wave & 3) << 4;    // t rows [strip, strip+16)

    v8f acc[4];
#pragma unroll
    for (int c = 0; c < 4; ++c)
#pragma unroll
        for (int r = 0; r < 8; ++r) acc[c][r] = bias[strip + 8 * h + r];

    const unsigned short* arow = &ldsW[(strip + l15) * KP];
#pragma unroll
    for (int s = 0; s < KK / 32; ++s) {
        const int kb = s * 32;
        FragBF a;
#pragma unroll
        for (int i = 0; i < 4; ++i) {
            a.u[i]     = *(const unsigned*)(arow + kb + 16 * h + 2 * i);
            a.u[4 + i] = *(const unsigned*)(arow + kb + 16 * h + 8 + 2 * i);
        }
        FragBF b[4];
#pragma unroll
        for (int c = 0; c < 4; ++c) {
            const unsigned short* brow = &ldsX[g][(c * 16 + l15) * KP];
#pragma unroll
            for (int v = 0; v < 8; ++v)
                b[c].u[v] = *(const unsigned*)(brow + kb + 16 * h + 2 * v);
        }
#pragma unroll
        for (int c = 0; c < 4; ++c)
            acc[c] = __builtin_amdgcn_wmma_f32_16x16x32_bf16(
                false, a.bf, false, b[c].bf, (short)0, acc[c], false, false);
    }

    const size_t base = (size_t)(node0 + g) * (TT * C_OUT);
#pragma unroll
    for (int c = 0; c < 4; ++c) {
        const int col = c * 16 + l15;
#pragma unroll
        for (int r = 0; r < 8; ++r) {
            const int row = strip + 8 * h + r;
            const size_t idx = base + (size_t)row * C_OUT + col;
            if (OUTBF) OutBf[idx] = f2bf(acc[c][r]);
            else       OutF[idx]  = acc[c][r];
        }
    }
}

// =====================================================================
// Big GEMM: Z_b = L (2048x2048, bf16) x Y_b (2048x4096, bf16)
//   MODE 0: O = acc;  MODE 1: O = 2*acc - T0
// 128x128 tile / block (8 waves), K-steps of 32, software-pipelined with
// double-buffered LDS: TDM DMA for A(ks+1) + global loads for B(ks+1)
// overlap the WMMA chain on tile ks. One barrier per K-step.
// =====================================================================
template <int MODE>
__global__ void gemm_kernel(const unsigned short* __restrict__ Lbf,
                            const unsigned short* __restrict__ Ybf,
                            const unsigned short* __restrict__ T0bf,
                            unsigned short* __restrict__ Obf) {
    constexpr int SA = 40;           // padded lds row (u16 units) == 80 bytes
    constexpr int KSTEPS = NN / 32;  // 64
    __shared__ unsigned short ldsA[2][128 * SA];
    __shared__ unsigned short ldsB[2][128 * SA];   // [j][k]

    const int tid = threadIdx.x;
    const int m0 = blockIdx.y * 128;
    const int j0 = blockIdx.x * 128;
    const size_t ybase = (size_t)blockIdx.z * NN * JCOLS;

    const int wave = tid >> 5, lane = tid & 31;
    const int h = lane >> 4, l15 = lane & 15;
    const int kr = tid >> 3, jseg = tid & 7;       // B-tile staging role
    const int ar = tid >> 1, aseg = tid & 1;       // A-tile staging role (fallback)

    v8f acc[8];
#pragma unroll
    for (int c = 0; c < 8; ++c)
#pragma unroll
        for (int r = 0; r < 8; ++r) acc[c][r] = 0.0f;

#if defined(USE_TDM)
    unsigned ldsaBase[2];
    ldsaBase[0] = (unsigned)(unsigned long long)(const void*)&ldsA[0][0];
    ldsaBase[1] = (unsigned)(unsigned long long)(const void*)&ldsA[1][0];
#endif

    // ---------------- prologue: stage tile 0 into buffer 0 ----------------
    {
        const unsigned short* g = Ybf + ybase + (size_t)kr * JCOLS + j0 + jseg * 16;
        uint4 b0 = *(const uint4*)g;
        uint4 b1 = *(const uint4*)(g + 8);
        unsigned short tmp[16];
        *(uint4*)tmp = b0; *(uint4*)(tmp + 8) = b1;
#pragma unroll
        for (int jj = 0; jj < 16; ++jj)
            ldsB[0][(jseg * 16 + jj) * SA + kr] = tmp[jj];
    }
#if defined(USE_TDM)
    if (tid < 32) {
        tdm_load_A(Lbf + (size_t)m0 * NN, ldsaBase[0]);
        __builtin_amdgcn_s_wait_tensorcnt(0);
    }
#elif defined(USE_ASYNC)
    {
        const unsigned short* g = Lbf + (size_t)(m0 + ar) * NN + aseg * 16;
        unsigned short* l = &ldsA[0][ar * SA + aseg * 16];
        __builtin_amdgcn_global_load_async_to_lds_b128(
            (const __attribute__((address_space(1))) void*)g,
            (__attribute__((address_space(3))) void*)l, 0, 0);
        __builtin_amdgcn_global_load_async_to_lds_b128(
            (const __attribute__((address_space(1))) void*)(g + 8),
            (__attribute__((address_space(3))) void*)(l + 8), 0, 0);
#  if __has_builtin(__builtin_amdgcn_s_wait_asynccnt)
        __builtin_amdgcn_s_wait_asynccnt(0);
#  endif
    }
#else
    {
        const unsigned short* g = Lbf + (size_t)(m0 + ar) * NN + aseg * 16;
        uint4 a0 = *(const uint4*)g;
        uint4 a1 = *(const uint4*)(g + 8);
        *(uint4*)&ldsA[0][ar * SA + aseg * 16]     = a0;
        *(uint4*)&ldsA[0][ar * SA + aseg * 16 + 8] = a1;
    }
#endif
    __syncthreads();

    // ---------------- main pipelined loop ----------------
    int p = 0;
    for (int ks = 0; ks < KSTEPS; ++ks) {
        const int kn = ks + 1;
        const bool pf = (kn < KSTEPS);
        uint4 b0 = {}, b1 = {};
#if !defined(USE_TDM) && !defined(USE_ASYNC)
        uint4 a0 = {}, a1 = {};
#endif
        if (pf) {
            // issue next-tile traffic before compute so it overlaps the WMMAs
            const unsigned short* g =
                Ybf + ybase + (size_t)(kn * 32 + kr) * JCOLS + j0 + jseg * 16;
            b0 = *(const uint4*)g;
            b1 = *(const uint4*)(g + 8);
#if defined(USE_TDM)
            if (tid < 32)
                tdm_load_A(Lbf + (size_t)m0 * NN + kn * 32, ldsaBase[p ^ 1]);
#elif defined(USE_ASYNC)
            {
                const unsigned short* ga = Lbf + (size_t)(m0 + ar) * NN + kn * 32 + aseg * 16;
                unsigned short* l = &ldsA[p ^ 1][ar * SA + aseg * 16];
                __builtin_amdgcn_global_load_async_to_lds_b128(
                    (const __attribute__((address_space(1))) void*)ga,
                    (__attribute__((address_space(3))) void*)l, 0, 0);
                __builtin_amdgcn_global_load_async_to_lds_b128(
                    (const __attribute__((address_space(1))) void*)(ga + 8),
                    (__attribute__((address_space(3))) void*)(l + 8), 0, 0);
            }
#else
            {
                const unsigned short* ga = Lbf + (size_t)(m0 + ar) * NN + kn * 32 + aseg * 16;
                a0 = *(const uint4*)ga;
                a1 = *(const uint4*)(ga + 8);
            }
#endif
        }

        // ---- compute on buffer p: 16x128 strip per wave
        const unsigned short* ldsAp = ldsA[p];
        const unsigned short* ldsBp = ldsB[p];
        FragBF a;
        const unsigned short* arow = &ldsAp[(wave * 16 + l15) * SA];
#pragma unroll
        for (int i = 0; i < 4; ++i) {
            a.u[i]     = *(const unsigned*)(arow + 16 * h + 2 * i);
            a.u[4 + i] = *(const unsigned*)(arow + 16 * h + 8 + 2 * i);
        }
        FragBF b[8];
#pragma unroll
        for (int c = 0; c < 8; ++c) {
            const unsigned short* brow = &ldsBp[(c * 16 + l15) * SA];
#pragma unroll
            for (int v = 0; v < 8; ++v)
                b[c].u[v] = *(const unsigned*)(brow + 16 * h + 2 * v);
        }
#pragma unroll
        for (int c = 0; c < 8; ++c)
            acc[c] = __builtin_amdgcn_wmma_f32_16x16x32_bf16(
                false, a.bf, false, b[c].bf, (short)0, acc[c], false, false);

        // ---- commit next tile into buffer p^1
        if (pf) {
            unsigned short tmp[16];
            *(uint4*)tmp = b0; *(uint4*)(tmp + 8) = b1;
            unsigned short* ldsBn = ldsB[p ^ 1];
#pragma unroll
            for (int jj = 0; jj < 16; ++jj)
                ldsBn[(jseg * 16 + jj) * SA + kr] = tmp[jj];
#if defined(USE_TDM)
            if (tid < 32) __builtin_amdgcn_s_wait_tensorcnt(0);
#elif defined(USE_ASYNC)
#  if __has_builtin(__builtin_amdgcn_s_wait_asynccnt)
            __builtin_amdgcn_s_wait_asynccnt(0);
#  endif
#else
            *(uint4*)&ldsA[p ^ 1][ar * SA + aseg * 16]     = a0;
            *(uint4*)&ldsA[p ^ 1][ar * SA + aseg * 16 + 8] = a1;
#endif
        }
        __syncthreads();
        p ^= 1;
    }

    // ---------------- epilogue ----------------
    const int mw = m0 + wave * 16;
#pragma unroll
    for (int c = 0; c < 8; ++c) {
        const int col = j0 + c * 16 + l15;
#pragma unroll
        for (int r = 0; r < 8; ++r) {
            const int row = mw + 8 * h + r;
            const size_t idx = ybase + (size_t)row * JCOLS + col;
            float v = acc[c][r];
            if (MODE == 1) v = 2.0f * v - bf2f(T0bf[idx]);
            Obf[idx] = f2bf(v);
        }
    }
}

// =====================================================================
// Chebyshev channel mix: Out[m,d] = sum_k [T0|T1|T2][m,k]*Wcat[k,d] + cb[d]
// =====================================================================
__global__ void cheb_kernel(const unsigned short* __restrict__ T0,
                            const unsigned short* __restrict__ T1,
                            const unsigned short* __restrict__ T2,
                            const unsigned short* __restrict__ Wcat,
                            const float* __restrict__ cb,
                            float* __restrict__ Out) {
    constexpr int SA = 40;
    constexpr int SW = 200;   // 192 + pad
    __shared__ unsigned short ldsA[128 * SA];
    __shared__ unsigned short ldsW[64 * SW];     // [d][k]

    const int tid = threadIdx.x;
    const size_t m0 = (size_t)blockIdx.x * 128;

    for (int i = tid; i < 192 * 64; i += 256) {
        int k = i / 64, d = i % 64;
        ldsW[d * SW + k] = Wcat[i];
    }

    const int wave = tid >> 5, lane = tid & 31;
    const int h = lane >> 4, l15 = lane & 15;

    v8f acc[4];
#pragma unroll
    for (int c = 0; c < 4; ++c) {
        const float bv = cb[c * 16 + l15];
#pragma unroll
        for (int r = 0; r < 8; ++r) acc[c][r] = bv;
    }

    const unsigned short* bufs[3] = {T0, T1, T2};
    for (int s = 0; s < 6; ++s) {
        __syncthreads();
        const unsigned short* src = bufs[s >> 1];
        const int inner = (s & 1) * 32;
        {
            const int r = tid >> 1, seg = tid & 1;
            const unsigned short* g = src + (m0 + r) * 64 + inner + seg * 16;
            uint4 v0 = *(const uint4*)g;
            uint4 v1 = *(const uint4*)(g + 8);
            *(uint4*)&ldsA[r * SA + seg * 16]     = v0;
            *(uint4*)&ldsA[r * SA + seg * 16 + 8] = v1;
        }
        __syncthreads();

        const int kb = s * 32;
        FragBF a;
        const unsigned short* arow = &ldsA[(wave * 16 + l15) * SA];
#pragma unroll
        for (int i = 0; i < 4; ++i) {
            a.u[i]     = *(const unsigned*)(arow + 16 * h + 2 * i);
            a.u[4 + i] = *(const unsigned*)(arow + 16 * h + 8 + 2 * i);
        }
        FragBF b[4];
#pragma unroll
        for (int c = 0; c < 4; ++c) {
            const unsigned short* brow = &ldsW[(c * 16 + l15) * SW + kb];
#pragma unroll
            for (int v = 0; v < 8; ++v)
                b[c].u[v] = *(const unsigned*)(brow + 16 * h + 2 * v);
        }
#pragma unroll
        for (int c = 0; c < 4; ++c)
            acc[c] = __builtin_amdgcn_wmma_f32_16x16x32_bf16(
                false, a.bf, false, b[c].bf, (short)0, acc[c], false, false);
    }

#pragma unroll
    for (int c = 0; c < 4; ++c) {
        const int col = c * 16 + l15;
#pragma unroll
        for (int r = 0; r < 8; ++r) {
            const size_t row = m0 + wave * 16 + 8 * h + r;
            Out[row * 64 + col] = acc[c][r];
        }
    }
}

// =====================================================================
// BatchNorm (training stats) + ReLU over [MTOT][64] in place on Y.
// =====================================================================
__global__ void bn_zero(float* __restrict__ stats) {
    if (threadIdx.x < 128) stats[threadIdx.x] = 0.0f;
}

__global__ void bn_stats(const float* __restrict__ Y, float* __restrict__ stats) {
    __shared__ float ssum[256], ssq[256];
    const int tid = threadIdx.x;
    const int c = tid & 63, grp = tid >> 6;
    const size_t rows = (size_t)MTOT / gridDim.x;
    const size_t r0 = (size_t)blockIdx.x * rows;
    float s = 0.0f, q = 0.0f;
    for (size_t r = r0 + grp; r < r0 + rows; r += 4) {
        float v = Y[r * 64 + c];
        s += v; q += v * v;
    }
    ssum[tid] = s; ssq[tid] = q;
    __syncthreads();
    if (grp == 0) {
        s = ssum[c] + ssum[64 + c] + ssum[128 + c] + ssum[192 + c];
        q = ssq[c]  + ssq[64 + c]  + ssq[128 + c]  + ssq[192 + c];
        atomicAdd(&stats[c], s);
        atomicAdd(&stats[64 + c], q);
    }
}

__global__ void bn_norm(float* __restrict__ Y, const float* __restrict__ stats,
                        const float* __restrict__ gamma, const float* __restrict__ beta) {
    const size_t i4 = (size_t)blockIdx.x * blockDim.x + threadIdx.x;
    const size_t n4 = (size_t)MTOT * 64 / 4;
    if (i4 >= n4) return;
    float4 v = ((float4*)Y)[i4];
    const int cbase = (int)((i4 * 4) & 63);
    float o[4] = {v.x, v.y, v.z, v.w};
#pragma unroll
    for (int j = 0; j < 4; ++j) {
        const int c = cbase + j;
        const float mean = stats[c] * (1.0f / (float)MTOT);
        const float var  = stats[64 + c] * (1.0f / (float)MTOT) - mean * mean;
        const float sc   = rsqrtf(var + EPSV) * gamma[c];
        float y = (o[j] - mean) * sc + beta[c];
        o[j] = y > 0.0f ? y : 0.0f;
    }
    ((float4*)Y)[i4] = make_float4(o[0], o[1], o[2], o[3]);
}

// =====================================================================
extern "C" void kernel_launch(void* const* d_in, const int* in_sizes, int n_in,
                              void* d_out, int out_size, void* d_ws, size_t ws_size,
                              hipStream_t stream) {
    const float* x     = (const float*)d_in[0];
    const float* Lt    = (const float*)d_in[1];
    const float* w1    = (const float*)d_in[2];
    const float* b1    = (const float*)d_in[3];
    const float* chw   = (const float*)d_in[4];
    const float* chb   = (const float*)d_in[5];
    const float* w2    = (const float*)d_in[6];
    const float* b2    = (const float*)d_in[7];
    const float* gamma = (const float*)d_in[8];
    const float* beta  = (const float*)d_in[9];
    float* out = (float*)d_out;

    // workspace layout
    char* w = (char*)d_ws;
    unsigned short* Lbf  = (unsigned short*)w;                 w += (size_t)NN * NN * 2;
    unsigned short* T0bf = (unsigned short*)w;                 w += (size_t)MTOT * 64 * 2;
    unsigned short* T1bf = (unsigned short*)w;                 w += (size_t)MTOT * 64 * 2;
    unsigned short* T2bf = (unsigned short*)w;                 w += (size_t)MTOT * 64 * 2;
    float*          ych  = (float*)w;                          w += (size_t)MTOT * 64 * 4;
    unsigned short* W1m  = (unsigned short*)w;                 w += 64 * 96 * 2 + 1024;
    unsigned short* W2m  = (unsigned short*)w;                 w += 64 * 192 * 2 + 1024;
    unsigned short* Wcat = (unsigned short*)w;                 w += 192 * 64 * 2 + 1024;
    float*          stats = (float*)w;

    {
        const long NL = (long)NN * NN;
        prep_kernel<<<(int)((NL + 255) / 256), 256, 0, stream>>>(
            Lt, w1, w2, chw, Lbf, W1m, W2m, Wcat);
    }
    tconv_kernel<C_IN, true><<<NODES / 2, 256, 0, stream>>>(x, W1m, b1, T0bf, nullptr);
    gemm_kernel<0><<<dim3(JCOLS / 128, NN / 128, BB), 256, 0, stream>>>(Lbf, T0bf, nullptr, T1bf);
    gemm_kernel<1><<<dim3(JCOLS / 128, NN / 128, BB), 256, 0, stream>>>(Lbf, T1bf, T0bf, T2bf);
    cheb_kernel<<<MTOT / 128, 256, 0, stream>>>(T0bf, T1bf, T2bf, Wcat, chb, ych);
    tconv_kernel<C_OUT, false><<<NODES / 2, 256, 0, stream>>>(ych, W2m, b2, nullptr, out);
    bn_zero<<<1, 128, 0, stream>>>(stats);
    bn_stats<<<4096, 256, 0, stream>>>(out, stats);
    bn_norm<<<(int)(((size_t)MTOT * 64 / 4 + 255) / 256), 256, 0, stream>>>(out, stats, gamma, beta);
}
